// AttentionBlock_41059887349938
// MI455X (gfx1250) — matrix operand order repaired and checked by reading, not compiled, not run on hardware
//
#include <hip/hip_runtime.h>

typedef __attribute__((ext_vector_type(16))) _Float16 v16h;
typedef __attribute__((ext_vector_type(8)))  _Float16 v8h;
typedef __attribute__((ext_vector_type(8)))  float    v8f;

#define B_N   4
#define S_LEN 4096
#define D_DIM 128
#define K_DIM 64
#define V_DIM 128

#define WMMA_F16(a, b, c) \
  __builtin_amdgcn_wmma_f32_16x16x32_f16(false, (a), false, (b), (short)0, (c), false, false)

// ---------------------------------------------------------------------------
// Fragment loaders (wave32, per ISA 7.12.2 layouts)
// A (16x32, f16): lane L holds row m=L&15; halves 0..7 -> K = kb..kb+7,
//                 halves 8..15 -> K = kb+16..kb+23, kb = (L<16 ? 0 : 8).
// B (32x16, f16) built as transpose-of-rows: B[k][n] = src[n0+n][kc+k];
//                 lane L holds col n=L&15, same K layout as A (kb, kb+16).
// Both load as two contiguous 16-byte v8h chunks per lane.
// ---------------------------------------------------------------------------
__device__ __forceinline__ v16h load_a_frag(const _Float16* base, int pitch,
                                            int row0, int kc) {
  const int lane = threadIdx.x & 31;
  const int m    = lane & 15;
  const int kb   = (lane < 16) ? 0 : 8;
  const _Float16* p = base + (size_t)(row0 + m) * pitch + kc + kb;
  v8h lo = *(const v8h*)p;
  v8h hi = *(const v8h*)(p + 16);
  v16h a;
#pragma unroll
  for (int h = 0; h < 8; ++h) { a[h] = lo[h]; a[8 + h] = hi[h]; }
  return a;
}

__device__ __forceinline__ v16h load_bt_frag(const _Float16* base, int pitch,
                                             int n0, int kc) {
  const int lane = threadIdx.x & 31;
  const int n    = lane & 15;
  const int kb   = (lane < 16) ? 0 : 8;
  const _Float16* p = base + (size_t)(n0 + n) * pitch + kc + kb;
  v8h lo = *(const v8h*)p;
  v8h hi = *(const v8h*)(p + 16);
  v16h b;
#pragma unroll
  for (int h = 0; h < 8; ++h) { b[h] = lo[h]; b[8 + h] = hi[h]; }
  return b;
}

// ---------------------------------------------------------------------------
// Kernel 0: pack x -> f16, weights -> packed f16 [256][128], biases -> [256]
// ---------------------------------------------------------------------------
__global__ __launch_bounds__(256)
void ab_pack(const float* __restrict__ x,
             const float* __restrict__ Wq, const float* __restrict__ bq,
             const float* __restrict__ Wk, const float* __restrict__ bk,
             const float* __restrict__ Wv, const float* __restrict__ bvec,
             _Float16* __restrict__ xh, _Float16* __restrict__ Wh,
             float* __restrict__ bAll) {
  const size_t i  = (size_t)blockIdx.x * blockDim.x + threadIdx.x;
  const size_t NX = (size_t)B_N * S_LEN * D_DIM;
  if (i < NX) xh[i] = (_Float16)x[i];
  if (i < 256u * 128u) {
    int r = (int)(i >> 7), c = (int)(i & 127);
    float w;
    if (r < 64)       w = Wq[r * 128 + c];
    else if (r < 128) w = Wk[(r - 64) * 128 + c];
    else              w = Wv[(r - 128) * 128 + c];
    Wh[i] = (_Float16)w;
  }
  if (i < 256) {
    float bb;
    if (i < 64)       bb = bq[i];
    else if (i < 128) bb = bk[i - 64];
    else              bb = bvec[i - 128];
    bAll[i] = bb;
  }
}

// ---------------------------------------------------------------------------
// Kernel 1: projections. One wave per 16-row tile. 16 WMMA n-tiles:
//   n-tiles 0..3 -> Q (f16), 4..7 -> K (f16), 8..15 -> V (f32).
// Also copies x into out[:, 0:128].
// ---------------------------------------------------------------------------
__global__ __launch_bounds__(128)
void ab_proj(const _Float16* __restrict__ xh, const float* __restrict__ x,
             const _Float16* __restrict__ Wh, const float* __restrict__ bAll,
             _Float16* __restrict__ Qh, _Float16* __restrict__ Kh,
             float* __restrict__ Vf, float* __restrict__ out) {
  const int wave = threadIdx.x >> 5;
  const int lane = threadIdx.x & 31;
  const int tile = blockIdx.x * 4 + wave;   // 0 .. B*S/16-1
  const int row0 = tile * 16;               // flat row over [B*S)

  v16h a0 = load_a_frag(xh, D_DIM, row0, 0);
  v16h a1 = load_a_frag(xh, D_DIM, row0, 32);
  v16h a2 = load_a_frag(xh, D_DIM, row0, 64);
  v16h a3 = load_a_frag(xh, D_DIM, row0, 96);

  const int n     = lane & 15;
  const int mbase = (lane < 16) ? 0 : 8;

  for (int nt = 0; nt < 16; ++nt) {
    const int n0 = nt * 16;
    v8f c = {};
    c = WMMA_F16(a0, load_bt_frag(Wh, D_DIM, n0, 0),  c);
    c = WMMA_F16(a1, load_bt_frag(Wh, D_DIM, n0, 32), c);
    c = WMMA_F16(a2, load_bt_frag(Wh, D_DIM, n0, 64), c);
    c = WMMA_F16(a3, load_bt_frag(Wh, D_DIM, n0, 96), c);
    const float bias = bAll[n0 + n];
    if (nt < 4) {
#pragma unroll
      for (int r = 0; r < 8; ++r)
        Qh[(size_t)(row0 + mbase + r) * K_DIM + n0 + n] = (_Float16)(c[r] + bias);
    } else if (nt < 8) {
#pragma unroll
      for (int r = 0; r < 8; ++r)
        Kh[(size_t)(row0 + mbase + r) * K_DIM + (n0 - 64) + n] = (_Float16)(c[r] + bias);
    } else {
#pragma unroll
      for (int r = 0; r < 8; ++r)
        Vf[(size_t)(row0 + mbase + r) * V_DIM + (n0 - 128) + n] = c[r] + bias;
    }
  }

  // concat part 1: out[:, 0:128] = x   (16 rows x 32 float4)
  for (int idx = lane; idx < 16 * 32; idx += 32) {
    const int rr = idx >> 5, c4 = idx & 31;
    float4 v = *(const float4*)(x + (size_t)(row0 + rr) * D_DIM + c4 * 4);
    *(float4*)(out + (size_t)(row0 + rr) * (D_DIM + V_DIM) + c4 * 4) = v;
  }
}

// ---------------------------------------------------------------------------
// Kernel 2: per (b, jTile) strip: colsum[j] = sum_{i>=j} exp(s[i,j]/8),
// then VscT[b][v][j] = V[b][j][v] / colsum[j]  (f16, transposed for PV B-frags)
// ---------------------------------------------------------------------------
__global__ __launch_bounds__(128)
void ab_colsum(const _Float16* __restrict__ Qh, const _Float16* __restrict__ Kh,
               const float* __restrict__ Vf, _Float16* __restrict__ VscT) {
  __shared__ float lds_colsum[16];
  const int b     = blockIdx.x >> 8;
  const int jTile = blockIdx.x & 255;
  const int wave  = threadIdx.x >> 5;
  const int lane  = threadIdx.x & 31;
  const int j0    = jTile * 16;
  const size_t rowBase = (size_t)b * S_LEN;
  const _Float16* Qb = Qh + rowBase * K_DIM;
  const _Float16* Kb = Kh + rowBase * K_DIM;

  if (threadIdx.x < 16) lds_colsum[threadIdx.x] = 0.0f;
  __syncthreads();

  const v16h bK0 = load_bt_frag(Kb, K_DIM, j0, 0);
  const v16h bK1 = load_bt_frag(Kb, K_DIM, j0, 32);

  const int n     = lane & 15;
  const int mbase = (lane < 16) ? 0 : 8;
  float colpart = 0.0f;

  for (int iTile = jTile + wave; iTile < S_LEN / 16; iTile += 4) {
    const int i0 = iTile * 16;
    v8f c = {};
    c = WMMA_F16(load_a_frag(Qb, K_DIM, i0, 0),  bK0, c);
    c = WMMA_F16(load_a_frag(Qb, K_DIM, i0, 32), bK1, c);
    const bool diag = (iTile == jTile);
#pragma unroll
    for (int r = 0; r < 8; ++r) {
      const float s = c[r] * 0.125f;              // 1/sqrt(K)
      const bool live = !diag || (n <= (mbase + r));
      colpart += live ? __expf(s) : 0.0f;
    }
  }
  atomicAdd(&lds_colsum[n], colpart);
  __syncthreads();

  for (int t = threadIdx.x; t < 16 * V_DIM; t += 128) {
    const int jj = t & 15, v = t >> 4;
    const float val = Vf[(rowBase + j0 + jj) * V_DIM + v] / lds_colsum[jj];
    VscT[((size_t)b * V_DIM + v) * S_LEN + j0 + jj] = (_Float16)val;
  }
}

// ---------------------------------------------------------------------------
// Kernel 3: per (b, iTile): attention rows. 2 waves: each computes one 16x16
// exp(score) half into LDS (only diagonal pair masked), then both WMMA the
// 16x32 P fragment against their 64 columns of VscT. Writes out[:, 128:256].
// ---------------------------------------------------------------------------
__global__ __launch_bounds__(64)
void ab_attn(const _Float16* __restrict__ Qh, const _Float16* __restrict__ Kh,
             const _Float16* __restrict__ VscT, float* __restrict__ out) {
  __shared__ __align__(16) _Float16 ldsP[16 * 32];
  const int b     = blockIdx.x >> 8;
  const int iTile = blockIdx.x & 255;
  const int wave  = threadIdx.x >> 5;
  const int lane  = threadIdx.x & 31;
  const int i0    = iTile * 16;
  const size_t rowBase = (size_t)b * S_LEN;
  const _Float16* Qb = Qh + rowBase * K_DIM;
  const _Float16* Kb = Kh + rowBase * K_DIM;
  const _Float16* Vb = VscT + (size_t)b * V_DIM * S_LEN;

  const v16h aQ0 = load_a_frag(Qb, K_DIM, i0, 0);
  const v16h aQ1 = load_a_frag(Qb, K_DIM, i0, 32);

  v8f cacc[4] = {{}, {}, {}, {}};
  const int n     = lane & 15;
  const int mbase = (lane < 16) ? 0 : 8;
  const int pMax  = iTile >> 1;

  for (int p = 0; p <= pMax; ++p) {
    const int jBase = p * 32;
    const int j0    = jBase + wave * 16;
    if (j0 <= i0 + 15) {
      v8f s = {};
      s = WMMA_F16(aQ0, load_bt_frag(Kb, K_DIM, j0, 0),  s);
      s = WMMA_F16(aQ1, load_bt_frag(Kb, K_DIM, j0, 32), s);
      const bool lastPair = (p == pMax);
#pragma unroll
      for (int r = 0; r < 8; ++r) {
        const float sv = s[r] * 0.125f;
        const bool live = !lastPair || ((j0 + n) <= (i0 + mbase + r));
        ldsP[(mbase + r) * 32 + wave * 16 + n] = (_Float16)(live ? __expf(sv) : 0.0f);
      }
    } else {  // fully masked half (iTile even, last pair, wave 1)
#pragma unroll
      for (int r = 0; r < 8; ++r)
        ldsP[(mbase + r) * 32 + wave * 16 + n] = (_Float16)0.0f;
    }
    __syncthreads();
    const v16h aP = load_a_frag(ldsP, 32, 0, 0);
#pragma unroll
    for (int t = 0; t < 4; ++t)
      cacc[t] = WMMA_F16(aP, load_bt_frag(Vb, S_LEN, wave * 64 + t * 16, jBase), cacc[t]);
    __syncthreads();
  }

#pragma unroll
  for (int t = 0; t < 4; ++t) {
    const int n0 = wave * 64 + t * 16;
#pragma unroll
    for (int r = 0; r < 8; ++r)
      out[(rowBase + i0 + mbase + r) * (D_DIM + V_DIM) + D_DIM + n0 + n] = cacc[t][r];
  }
}

// ---------------------------------------------------------------------------
extern "C" void kernel_launch(void* const* d_in, const int* in_sizes, int n_in,
                              void* d_out, int out_size, void* d_ws, size_t ws_size,
                              hipStream_t stream) {
  (void)in_sizes; (void)n_in; (void)out_size; (void)ws_size;
  const float* x  = (const float*)d_in[0];
  const float* Wq = (const float*)d_in[1];
  const float* bq = (const float*)d_in[2];
  const float* Wk = (const float*)d_in[3];
  const float* bk = (const float*)d_in[4];
  const float* Wv = (const float*)d_in[5];
  const float* bv = (const float*)d_in[6];
  float* out = (float*)d_out;

  char* ws = (char*)d_ws;
  size_t off = 0;
  auto carve = [&](size_t bytes) -> void* {
    void* p = ws + off;
    off = (off + bytes + 255) & ~(size_t)255;
    return p;
  };
  const size_t NR = (size_t)B_N * S_LEN;          // 16384 rows
  _Float16* xh   = (_Float16*)carve(NR * D_DIM * sizeof(_Float16)); // 4 MB
  _Float16* Wh   = (_Float16*)carve(256 * 128 * sizeof(_Float16));  // 64 KB
  float*    bAll = (float*)   carve(256 * sizeof(float));           // 1 KB
  _Float16* Qh   = (_Float16*)carve(NR * K_DIM * sizeof(_Float16)); // 2 MB
  _Float16* Kh   = (_Float16*)carve(NR * K_DIM * sizeof(_Float16)); // 2 MB
  float*    Vf   = (float*)   carve(NR * V_DIM * sizeof(float));    // 8 MB
  _Float16* VscT = (_Float16*)carve(NR * V_DIM * sizeof(_Float16)); // 4 MB

  const size_t NX = NR * D_DIM;
  ab_pack<<<(unsigned)((NX + 255) / 256), 256, 0, stream>>>(
      x, Wq, bq, Wk, bk, Wv, bv, xh, Wh, bAll);
  ab_proj<<<(unsigned)(NR / 16 / 4), 128, 0, stream>>>(
      xh, x, Wh, bAll, Qh, Kh, Vf, out);
  ab_colsum<<<B_N * (S_LEN / 16), 128, 0, stream>>>(Qh, Kh, Vf, VscT);
  ab_attn<<<B_N * (S_LEN / 16), 64, 0, stream>>>(Qh, Kh, VscT, out);
}
